// TextSA_50251117363546
// MI455X (gfx1250) — matrix operand order repaired and checked by reading, not compiled, not run on hardware
//
#include <hip/hip_runtime.h>

// Shapes from the reference
#define BB 256
#define LL 128
#define DD 1024
#define PITCH 1032   // bf16 elements per LDS row (8-element pad -> bank spread)

typedef __attribute__((ext_vector_type(8)))  __bf16 v8bf;
typedef __attribute__((ext_vector_type(16))) __bf16 v16bf;
typedef __attribute__((ext_vector_type(8)))  float  v8f;

// fp32 -> bf16 with round-to-nearest-even (pure integer ops; always lowers)
__device__ __forceinline__ unsigned short f2bfu(float f) {
  unsigned u = __builtin_bit_cast(unsigned, f);
  unsigned r = u + 0x7FFFu + ((u >> 16) & 1u);
  return (unsigned short)(r >> 16);
}

// branch-free fast tanh: 1 - 2/(e^{2x}+1).  exp overflow -> rcp(inf)=0 -> 1;
// exp underflow -> 1-2 = -1.  No exec-mask divergence (v_exp + v_rcp + 3 VALU).
__device__ __forceinline__ float fast_tanh(float x) {
  float ez = __expf(2.0f * x);
  return 1.0f - 2.0f * __builtin_amdgcn_rcpf(ez + 1.0f);
}

// ---------------------------------------------------------------------------
// Kernel 1: convert Wl (1024x1024 fp32, row-major [e][d]) to bf16 workspace.
// Wl_bf16 stays hot in the 192MB L2 and is streamed by every score block.
// ---------------------------------------------------------------------------
__global__ void __launch_bounds__(256)
cvt_wl_kernel(const float4* __restrict__ Wl4, unsigned long long* __restrict__ WlB) {
  int idx = blockIdx.x * 256 + threadIdx.x;          // 262144 float4's
  float4 f = Wl4[idx];
  unsigned long long p =  (unsigned long long)f2bfu(f.x)
                       | ((unsigned long long)f2bfu(f.y) << 16)
                       | ((unsigned long long)f2bfu(f.z) << 32)
                       | ((unsigned long long)f2bfu(f.w) << 48);
  WlB[idx] = p;
}

// ---------------------------------------------------------------------------
// Kernel 2: per caption b: masked mean over valid words -> cap_ave,
// g_emb = tanh(cap_ave @ Wg^T + bg), fold scorer: v[b,e] = g_emb[b,e]*Wc[e].
// 0.54 GFLOP total -> scalar VALU is fine.
// ---------------------------------------------------------------------------
__global__ void __launch_bounds__(256)
prep_kernel(const float* __restrict__ local, const int* __restrict__ lengths,
            const float* __restrict__ Wg, const float* __restrict__ bg,
            const float* __restrict__ Wc, float* __restrict__ vout) {
  __shared__ float cap[DD];
  const int tid = threadIdx.x;
  const int b   = blockIdx.x;
  const int len = lengths[b];

  const float4* src = (const float4*)(local + (size_t)b * LL * DD);
  float4 acc = make_float4(0.f, 0.f, 0.f, 0.f);
  for (int l = 0; l < len; ++l) {               // masked sum == sum over first len rows
    float4 f = src[l * (DD / 4) + tid];
    acc.x += f.x; acc.y += f.y; acc.z += f.z; acc.w += f.w;
  }
  const float inv = 1.0f / (float)len;
  cap[tid * 4 + 0] = acc.x * inv;
  cap[tid * 4 + 1] = acc.y * inv;
  cap[tid * 4 + 2] = acc.z * inv;
  cap[tid * 4 + 3] = acc.w * inv;
  __syncthreads();

  #pragma unroll
  for (int j = 0; j < 4; ++j) {
    const int e = tid + 256 * j;
    const float4* wr = (const float4*)(Wg + (size_t)e * DD);
    float s = 0.f;
    for (int k4 = 0; k4 < DD / 4; ++k4) {       // cap[] reads are lane-uniform (LDS broadcast)
      float4 w = wr[k4];
      s += w.x * cap[k4 * 4 + 0] + w.y * cap[k4 * 4 + 1]
         + w.z * cap[k4 * 4 + 2] + w.w * cap[k4 * 4 + 3];
    }
    float g = fast_tanh(s + bg[e]);
    vout[(size_t)b * DD + e] = g * Wc[e];
  }
}

// ---------------------------------------------------------------------------
// Kernel 3 (main): per caption b, fused  scores[l] = sum_e tanh(local[b]@Wl^T)[l,e]*v[b,e]
// then masked softmax. 8 wave32 per block; whole bf16 local[b] panel (258KB) in LDS.
// GEMM per block: M=128, N=1024, K=1024 via v_wmma_f32_16x16x32_bf16.
// B-fragment is loaded ONCE per (nt,k) and fed to 8 independent accumulator
// chains (all 8 M-tiles), cutting L2 weight traffic 8x and hiding WMMA latency.
// ---------------------------------------------------------------------------
__global__ void __launch_bounds__(256)
score_kernel(const float* __restrict__ local, const int* __restrict__ lengths,
             const unsigned short* __restrict__ WlB, const float* __restrict__ bl,
             const float* __restrict__ vin, const float* __restrict__ bc,
             float* __restrict__ out) {
  extern __shared__ char smem[];
  unsigned short* Abf    = (unsigned short*)smem;                    // 128 x PITCH bf16
  float*          sScore = (float*)(smem + (size_t)LL * PITCH * 2);  // 128 floats
  float*          redA   = sScore + LL;                              // 128 floats

  const int tid = threadIdx.x;
  const int b   = blockIdx.x;

  // ---- stage local[b] fp32 -> bf16 rows in LDS (each thread: one float4/row) ----
  const float4* src = (const float4*)(local + (size_t)b * LL * DD);
  for (int row = 0; row < LL; ++row) {
    float4 f = src[row * (DD / 4) + tid];
    unsigned long long p =  (unsigned long long)f2bfu(f.x)
                         | ((unsigned long long)f2bfu(f.y) << 16)
                         | ((unsigned long long)f2bfu(f.z) << 32)
                         | ((unsigned long long)f2bfu(f.w) << 48);
    *(unsigned long long*)(Abf + row * PITCH + tid * 4) = p;
  }
  if (tid < LL) sScore[tid] = 0.f;
  __syncthreads();

  const int wave = tid >> 5;
  const int lane = tid & 31;
  const int lm   = lane & 15;   // A: row-in-tile; B/C: column-in-tile
  const int h    = lane >> 4;   // lane half selects K sub-range / C row offset

  // each wave owns 128 output columns: e in [wave*128, wave*128+128)
  for (int nt = 0; nt < 8; ++nt) {
    const int   e    = wave * 128 + nt * 16 + lm;
    const float blv  = bl[e];
    const float vv   = vin[(size_t)b * DD + e];
    const unsigned short* wrow = WlB + (size_t)e * DD;   // Wl[e][:] as bf16 (K contiguous)
    __builtin_prefetch(wrow, 0, 1);                      // -> global_prefetch_b8

    v8f acc[8] = {};   // one 16x16 accumulator tile per M-tile (all 8 rows of L)
    const unsigned short* arow0 = Abf + lm * PITCH + 16 * h;

    for (int k = 0; k < DD; k += 32) {
      // B-fragment: lane (n + 16h) holds column n, K = k+16h .. k+16h+15 (contiguous)
      v16bf bfr = *(const v16bf*)(wrow + k + 16 * h);
      #pragma unroll
      for (int mt = 0; mt < 8; ++mt) {
        // A-fragment (matches B): lanes 0-15 hold K {k..k+15},
        //                         lanes 16-31 hold K {k+16..k+31}
        const unsigned short* ar = arow0 + (mt * 16) * PITCH + k;
        v8bf alo = *(const v8bf*)(ar);
        v8bf ahi = *(const v8bf*)(ar + 8);
        v16bf af = __builtin_shufflevector(alo, ahi,
                     0, 1, 2, 3, 4, 5, 6, 7, 8, 9, 10, 11, 12, 13, 14, 15);
        acc[mt] = __builtin_amdgcn_wmma_f32_16x16x32_bf16(
                      false, af, false, bfr, (short)0, acc[mt], false, false);
      }
    }

    // epilogue: acc[mt][j] = D[m = j+8h][n = lm]; tanh, scale by v[b,e], reduce over n
    #pragma unroll
    for (int mt = 0; mt < 8; ++mt) {
      #pragma unroll
      for (int j = 0; j < 8; ++j) {
        float t = fast_tanh(acc[mt][j] + blv) * vv;
        t += __shfl_xor(t, 1, 32);
        t += __shfl_xor(t, 2, 32);
        t += __shfl_xor(t, 4, 32);
        t += __shfl_xor(t, 8, 32);      // stays within the 16-lane half
        if (lm == 0) atomicAdd(&sScore[mt * 16 + j + 8 * h], t);  // ds_add_f32
      }
    }
  }
  __syncthreads();

  // ---- masked softmax over the 128 positions ----
  const int  len   = lengths[b];
  const bool inr   = tid < LL;
  const bool valid = tid < len;
  float sc = valid ? (sScore[tid] + bc[0]) : -3.4e38f;
  if (inr) redA[tid] = sc;
  __syncthreads();
  for (int s = LL / 2; s > 0; s >>= 1) {
    if (tid < s) redA[tid] = fmaxf(redA[tid], redA[tid + s]);
    __syncthreads();
  }
  const float mx = redA[0];
  __syncthreads();
  const float ex = valid ? __expf(sc - mx) : 0.f;
  if (inr) redA[tid] = ex;
  __syncthreads();
  for (int s = LL / 2; s > 0; s >>= 1) {
    if (tid < s) redA[tid] += redA[tid + s];
    __syncthreads();
  }
  const float ssum = redA[0];
  if (inr) out[(size_t)b * LL + tid] = valid ? (ex / ssum) : 0.f;
}

// ---------------------------------------------------------------------------
extern "C" void kernel_launch(void* const* d_in, const int* in_sizes, int n_in,
                              void* d_out, int out_size, void* d_ws, size_t ws_size,
                              hipStream_t stream) {
  (void)in_sizes; (void)n_in; (void)out_size; (void)ws_size;
  const float* local   = (const float*)d_in[0];
  const int*   lengths = (const int*)  d_in[1];
  const float* Wl      = (const float*)d_in[2];
  const float* bl      = (const float*)d_in[3];
  const float* Wg      = (const float*)d_in[4];
  const float* bg      = (const float*)d_in[5];
  const float* Wc      = (const float*)d_in[6];
  const float* bc      = (const float*)d_in[7];
  float* out = (float*)d_out;

  // workspace: v[B*D] fp32 (1MB) | Wl_bf16[D*D] (2MB)
  float*          v   = (float*)d_ws;
  unsigned short* WlB = (unsigned short*)((char*)d_ws + (size_t)BB * DD * sizeof(float));

  cvt_wl_kernel<<<dim3((DD * DD / 4) / 256), dim3(256), 0, stream>>>(
      (const float4*)Wl, (unsigned long long*)WlB);
  prep_kernel<<<dim3(BB), dim3(256), 0, stream>>>(local, lengths, Wg, bg, Wc, v);

  const size_t smem_bytes = (size_t)LL * PITCH * 2 + 2 * LL * sizeof(float);
  score_kernel<<<dim3(BB), dim3(256), smem_bytes, stream>>>(
      local, lengths, WlB, bl, v, bc, out);
}
